// VisualRVQ_85091892068796
// MI455X (gfx1250) — matrix operand order repaired and checked by reading, not compiled, not run on hardware
//
#include <hip/hip_runtime.h>

typedef _Float16 v16h __attribute__((ext_vector_type(16)));
typedef _Float16 v8h  __attribute__((ext_vector_type(8)));
typedef _Float16 v4h  __attribute__((ext_vector_type(4)));
typedef float    v8f  __attribute__((ext_vector_type(8)));
typedef int      v4i  __attribute__((ext_vector_type(4)));

#define RVQ_D 768
#define RVQ_K 4096

#if defined(__HIP_DEVICE_COMPILE__) && __has_builtin(__builtin_amdgcn_global_load_async_to_lds_b128)
#define RVQ_HAVE_ASYNC 1
#else
#define RVQ_HAVE_ASYNC 0
#endif

#if RVQ_HAVE_ASYNC
__device__ __forceinline__ void rvq_wait_async0() {
#if __has_builtin(__builtin_amdgcn_s_wait_asynccnt)
  __builtin_amdgcn_s_wait_asynccnt(0);
#else
  asm volatile("s_wait_asynccnt 0x0" ::: "memory");
#endif
}
// Stage 32 codewords x 768 f16 (48 KB) : 12 x 16B per thread, no VGPR data path.
__device__ __forceinline__ void rvq_stage_async(const _Float16* gsrc,
                                                _Float16* ldst, int tid) {
  char* g = (char*)gsrc + tid * 16;
  char* l = (char*)ldst + tid * 16;
#pragma unroll
  for (int i = 0; i < 12; ++i)
    __builtin_amdgcn_global_load_async_to_lds_b128(
        (__attribute__((address_space(1))) v4i*)(g + i * 4096),
        (__attribute__((address_space(3))) v4i*)(l + i * 4096),
        0, 0);
}
#endif

// ---------------------------------------------------------------------------
// residual = features - quant_out ; rn = l2norm(residual) stored as f16
// ---------------------------------------------------------------------------
__global__ __launch_bounds__(256) void rvq_resid_norm(
    const float* __restrict__ feat, const float* __restrict__ quant,
    _Float16* __restrict__ rn)
{
  const int row  = (blockIdx.x * blockDim.x + threadIdx.x) >> 5;
  const int lane = threadIdx.x & 31;
  const float4* f4 = (const float4*)(feat  + (size_t)row * RVQ_D);
  const float4* q4 = (const float4*)(quant + (size_t)row * RVQ_D);
  float4 v[6]; float ss = 0.f;
#pragma unroll
  for (int i = 0; i < 6; ++i) {
    float4 a = f4[i * 32 + lane], b = q4[i * 32 + lane];
    v[i].x = a.x - b.x; v[i].y = a.y - b.y; v[i].z = a.z - b.z; v[i].w = a.w - b.w;
    ss += v[i].x * v[i].x + v[i].y * v[i].y + v[i].z * v[i].z + v[i].w * v[i].w;
  }
#pragma unroll
  for (int m = 16; m >= 1; m >>= 1) ss += __shfl_xor(ss, m, 32);
  const float s = rsqrtf(ss + 1e-12f);
  v4h* out = (v4h*)(rn + (size_t)row * RVQ_D);
#pragma unroll
  for (int i = 0; i < 6; ++i) {
    v4h h;
    h[0] = (_Float16)(v[i].x * s); h[1] = (_Float16)(v[i].y * s);
    h[2] = (_Float16)(v[i].z * s); h[3] = (_Float16)(v[i].w * s);
    out[i * 32 + lane] = h;
  }
}

// ---------------------------------------------------------------------------
// cbn = l2norm(codebooks[q]) stored as f16 [4096][768]
// ---------------------------------------------------------------------------
__global__ __launch_bounds__(256) void rvq_cb_norm(
    const float* __restrict__ cbq, _Float16* __restrict__ cbn)
{
  const int row  = (blockIdx.x * blockDim.x + threadIdx.x) >> 5;
  const int lane = threadIdx.x & 31;
  const float4* s4 = (const float4*)(cbq + (size_t)row * RVQ_D);
  float4 v[6]; float ss = 0.f;
#pragma unroll
  for (int i = 0; i < 6; ++i) {
    v[i] = s4[i * 32 + lane];
    ss += v[i].x * v[i].x + v[i].y * v[i].y + v[i].z * v[i].z + v[i].w * v[i].w;
  }
#pragma unroll
  for (int m = 16; m >= 1; m >>= 1) ss += __shfl_xor(ss, m, 32);
  const float s = rsqrtf(ss + 1e-12f);
  v4h* out = (v4h*)(cbn + (size_t)row * RVQ_D);
#pragma unroll
  for (int i = 0; i < 6; ++i) {
    v4h h;
    h[0] = (_Float16)(v[i].x * s); h[1] = (_Float16)(v[i].y * s);
    h[2] = (_Float16)(v[i].z * s); h[3] = (_Float16)(v[i].w * s);
    out[i * 32 + lane] = h;
  }
}

// ---------------------------------------------------------------------------
// sim = rn @ cbn^T fused with argmax over K=4096.
// Block: 8 waves x 16 rows. A tile resident in VGPRs; codebook streamed as
// 32-codeword tiles through double-buffered LDS via async-to-LDS loads.
// ---------------------------------------------------------------------------
__global__ __launch_bounds__(256) void rvq_sim_argmax(
    const _Float16* __restrict__ rn, const _Float16* __restrict__ cbn,
    int* __restrict__ idx_ws, float* __restrict__ out_idx, int qstep, int Q)
{
  extern __shared__ _Float16 lds[];          // 2 x 32*768 halves (96 KB)
  const int tid  = threadIdx.x;
  const int lane = tid & 31;
  const int wave = tid >> 5;
  const int rowBase = blockIdx.x * 128 + wave * 16;

  // ---- A fragments (f16 16x32 layout): row = lane&15,
  //      lane<16 holds K{0..15}, lane>=16 holds K{16..31} ----
  const _Float16* arow = rn + (size_t)(rowBase + (lane & 15)) * RVQ_D;
  const int klo = (lane < 16) ? 0 : 16;
  v8h a_lo[24], a_hi[24];
#pragma unroll
  for (int c = 0; c < 24; ++c) {
    a_lo[c] = *(const v8h*)(arow + c * 32 + klo);
    a_hi[c] = *(const v8h*)(arow + c * 32 + klo + 8);
  }

  float best[8]; int bidx[8];
#pragma unroll
  for (int r = 0; r < 8; ++r) { best[r] = -3.4e38f; bidx[r] = 0; }

  // ---- stage tile-pair 0 ----
#if RVQ_HAVE_ASYNC
  rvq_stage_async(cbn, lds, tid);
  rvq_wait_async0();
#else
  {
    v8h sreg[12];
    const v8h* src = (const v8h*)cbn;
#pragma unroll
    for (int i = 0; i < 12; ++i) sreg[i] = src[tid + i * 256];
    v8h* dst = (v8h*)lds;
#pragma unroll
    for (int i = 0; i < 12; ++i) dst[tid + i * 256] = sreg[i];
  }
#endif
  __syncthreads();

  const int nlane = lane & 15;
  const int kb = (lane < 16) ? 0 : 16;      // B 32x16: contiguous K run per lane

  for (int t = 0; t < RVQ_K / 32; ++t) {
    const _Float16* buf = lds + (size_t)(t & 1) * (32 * RVQ_D);
#if RVQ_HAVE_ASYNC
    if (t + 1 < RVQ_K / 32)
      rvq_stage_async(cbn + (size_t)(t + 1) * 32 * RVQ_D,
                      lds + (size_t)((t + 1) & 1) * (32 * RVQ_D), tid);
#else
    v8h sreg[12];
    if (t + 1 < RVQ_K / 32) {
      const v8h* src = (const v8h*)(cbn + (size_t)(t + 1) * 32 * RVQ_D);
#pragma unroll
      for (int i = 0; i < 12; ++i) sreg[i] = src[tid + i * 256];
    }
#endif

    v8f acc0 = {}; v8f acc1 = {};
    const _Float16* b0 = buf + (size_t)nlane * RVQ_D + kb;
    const _Float16* b1 = buf + (size_t)(nlane + 16) * RVQ_D + kb;
    // software-pipelined B fragments: load c+1 while WMMA consumes c
    v16h b0c = *(const v16h*)(b0);
    v16h b1c = *(const v16h*)(b1);
#pragma unroll
    for (int c = 0; c < 24; ++c) {
      v16h b0n = b0c, b1n = b1c;
      if (c < 23) {
        b0n = *(const v16h*)(b0 + (c + 1) * 32);
        b1n = *(const v16h*)(b1 + (c + 1) * 32);
      }
      v16h a = __builtin_shufflevector(a_lo[c], a_hi[c],
                 0,1,2,3,4,5,6,7,8,9,10,11,12,13,14,15);
      acc0 = __builtin_amdgcn_wmma_f32_16x16x32_f16(false, a, false, b0c,
                                                    (short)0, acc0, false, false);
      acc1 = __builtin_amdgcn_wmma_f32_16x16x32_f16(false, a, false, b1c,
                                                    (short)0, acc1, false, false);
      b0c = b0n; b1c = b1n;
    }

    const int n0 = t * 32 + nlane;          // C layout: lane = N, VGPR r = M
#pragma unroll
    for (int r = 0; r < 8; ++r) {
      if (acc0[r] > best[r]) { best[r] = acc0[r]; bidx[r] = n0; }
      if (acc1[r] > best[r]) { best[r] = acc1[r]; bidx[r] = n0 + 16; }
    }

#if RVQ_HAVE_ASYNC
    rvq_wait_async0();                      // staged data fully in LDS
#else
    if (t + 1 < RVQ_K / 32) {
      v8h* dst = (v8h*)(lds + (size_t)((t + 1) & 1) * (32 * RVQ_D));
#pragma unroll
      for (int i = 0; i < 12; ++i) dst[tid + i * 256] = sreg[i];
    }
#endif
    __syncthreads();
  }

  // ---- cross-lane argmax within each 16-lane half (ties -> lower index) ----
#pragma unroll
  for (int m = 8; m >= 1; m >>= 1) {
#pragma unroll
    for (int r = 0; r < 8; ++r) {
      float ov = __shfl_xor(best[r], m, 32);
      int   oi = __shfl_xor(bidx[r], m, 32);
      if (ov > best[r] || (ov == best[r] && oi < bidx[r])) { best[r] = ov; bidx[r] = oi; }
    }
  }
  if (nlane == 0) {
    const int mrow = rowBase + ((lane >> 4) << 3);
#pragma unroll
    for (int r = 0; r < 8; ++r) {
      idx_ws[mrow + r] = bidx[r];
      out_idx[(size_t)(mrow + r) * Q + qstep] = (float)bidx[r];
    }
  }
}

// ---------------------------------------------------------------------------
// quant_out += l2norm(codebooks[q][idx]) gathered in fp32 (exact semantics)
// ---------------------------------------------------------------------------
__global__ __launch_bounds__(256) void rvq_update(
    const float* __restrict__ cbq, const int* __restrict__ idx_ws,
    float* __restrict__ quant)
{
  const int row  = (blockIdx.x * blockDim.x + threadIdx.x) >> 5;
  const int lane = threadIdx.x & 31;
  const int idx  = idx_ws[row];
  const float4* s4 = (const float4*)(cbq + (size_t)idx * RVQ_D);
  float4 v[6]; float ss = 0.f;
#pragma unroll
  for (int i = 0; i < 6; ++i) {
    v[i] = s4[i * 32 + lane];
    ss += v[i].x * v[i].x + v[i].y * v[i].y + v[i].z * v[i].z + v[i].w * v[i].w;
  }
#pragma unroll
  for (int m = 16; m >= 1; m >>= 1) ss += __shfl_xor(ss, m, 32);
  const float s = rsqrtf(ss + 1e-12f);
  float4* q4 = (float4*)(quant + (size_t)row * RVQ_D);
#pragma unroll
  for (int i = 0; i < 6; ++i) {
    float4 o = q4[i * 32 + lane];
    o.x += v[i].x * s; o.y += v[i].y * s; o.z += v[i].z * s; o.w += v[i].w * s;
    q4[i * 32 + lane] = o;
  }
}

extern "C" void kernel_launch(void* const* d_in, const int* in_sizes, int n_in,
                              void* d_out, int out_size, void* d_ws, size_t ws_size,
                              hipStream_t stream) {
  const float* feat = (const float*)d_in[0];
  const float* cb   = (const float*)d_in[1];
  const int N = in_sizes[0] / RVQ_D;                    // 32768
  const int Q = in_sizes[1] / (RVQ_K * RVQ_D);          // 64

  float* quant   = (float*)d_out;
  float* out_idx = (float*)d_out + (size_t)N * RVQ_D;

  _Float16* rn  = (_Float16*)d_ws;                                   // 48 MB
  _Float16* cbn = rn + (size_t)N * RVQ_D;                            //  6 MB
  int* idx_ws   = (int*)(cbn + (size_t)RVQ_K * RVQ_D);               // 128 KB

  (void)hipMemsetAsync(quant, 0, (size_t)N * RVQ_D * sizeof(float), stream);

  const size_t lds_bytes = 2u * 32u * RVQ_D * sizeof(_Float16);      // 96 KB
  for (int q = 0; q < Q; ++q) {
    const float* cbq = cb + (size_t)q * RVQ_K * RVQ_D;
    rvq_resid_norm<<<N / 8, 256, 0, stream>>>(feat, quant, rn);
    rvq_cb_norm  <<<RVQ_K / 8, 256, 0, stream>>>(cbq, cbn);
    rvq_sim_argmax<<<N / 128, 256, lds_bytes, stream>>>(rn, cbn, idx_ws,
                                                        out_idx, q, Q);
    rvq_update   <<<N / 8, 256, 0, stream>>>(cbq, idx_ws, quant);
  }
}
